// ACDEncoderDecoder_67035849556365
// MI455X (gfx1250) — compile-verified, matrix operand order checked by reading, not run on hardware
//
#include <hip/hip_runtime.h>

// ---------------------------------------------------------------------------
// ACD encoder/decoder for MI455X (gfx1250, wave32, WMMA bf16 16x16x32)
// B=512, L=32, V=64, H=128
// ---------------------------------------------------------------------------

typedef __attribute__((ext_vector_type(16))) __bf16 v16bf;
typedef __attribute__((ext_vector_type(8)))  float  v8f;
typedef __attribute__((ext_vector_type(8)))  __bf16 bf16x8;
typedef __attribute__((ext_vector_type(4)))  float  f32x4;

#define NB   512
#define NL   32
#define NV   64
#define NH   128
#define K1   2048          // NV*NL
#define N2   131072        // NL*NV*NV

__device__ __forceinline__ v8f wmma_bf16(v16bf a, v16bf b, v8f c) {
    return __builtin_amdgcn_wmma_f32_16x16x32_bf16(false, a, false, b,
                                                   (short)0, c, false, false);
}

// A/B fragment of a 16x32 bf16 tile from a row-major matrix row pointer.
// Per lane: two contiguous 8-element K-runs -> two 16B loads.
__device__ __forceinline__ v16bf load_frag_bf16(const __bf16* row, int kc, int hi) {
    bf16x8 lo = *reinterpret_cast<const bf16x8*>(row + kc + hi * 8);
    bf16x8 hv = *reinterpret_cast<const bf16x8*>(row + kc + 16 + hi * 8);
    v16bf r;
#pragma unroll
    for (int i = 0; i < 8; ++i) { r[i] = lo[i]; r[i + 8] = hv[i]; }
    return r;
}

// Same fragment, sourced from f32 memory with in-register bf16 conversion.
__device__ __forceinline__ v16bf load_frag_f32(const float* row, int kc, int hi) {
    f32x4 a0 = *reinterpret_cast<const f32x4*>(row + kc + hi * 8);
    f32x4 a1 = *reinterpret_cast<const f32x4*>(row + kc + hi * 8 + 4);
    f32x4 b0 = *reinterpret_cast<const f32x4*>(row + kc + 16 + hi * 8);
    f32x4 b1 = *reinterpret_cast<const f32x4*>(row + kc + 16 + hi * 8 + 4);
    v16bf r;
#pragma unroll
    for (int i = 0; i < 4; ++i) {
        r[i]      = (__bf16)a0[i];
        r[i + 4]  = (__bf16)a1[i];
        r[i + 8]  = (__bf16)b0[i];
        r[i + 12] = (__bf16)b1[i];
    }
    return r;
}

__device__ __forceinline__ float fsigmoid(float x) {
    return 1.0f / (1.0f + __expf(-x));
}
__device__ __forceinline__ float ftanh(float x) {
    float e = __expf(2.0f * x);
    return (e - 1.0f) / (e + 1.0f);
}

// ---------------------------------------------------------------------------
// k_prep: bf16-convert recurrent weights, combine LSTM biases
// ---------------------------------------------------------------------------
__global__ __launch_bounds__(256) void k_prep(
    const float* __restrict__ Wih, const float* __restrict__ Whh,
    const float* __restrict__ Wd,  const float* __restrict__ bih,
    const float* __restrict__ bhh,
    __bf16* __restrict__ Wih_bf, __bf16* __restrict__ Whh_bf,
    __bf16* __restrict__ Wd_bf,  float* __restrict__ bcomb) {
    int id = blockIdx.x * 256 + threadIdx.x;
    if (id < 4 * NH * NV) Wih_bf[id] = (__bf16)Wih[id];
    if (id < 4 * NH * NH) Whh_bf[id] = (__bf16)Whh[id];
    if (id < NV * NH)     Wd_bf[id]  = (__bf16)Wd[id];
    if (id < 4 * NH)      bcomb[id]  = bih[id] + bhh[id];
}

// ---------------------------------------------------------------------------
// k_fc1: h = relu(x_flat @ W1^T + b1)  -> h_bf (512x128 bf16)
// grid: 32 blocks (M tiles), 8 waves = 8 N tiles
// ---------------------------------------------------------------------------
__global__ __launch_bounds__(256) void k_fc1(
    const float* __restrict__ x, const float* __restrict__ W1,
    const float* __restrict__ b1, __bf16* __restrict__ h_bf) {
    const int mtile = blockIdx.x;
    const int wv    = threadIdx.x >> 5;   // ntile 0..7
    const int lane  = threadIdx.x & 31;
    const int l16   = lane & 15;
    const int hi    = lane >> 4;

    const float* arow = x  + (size_t)(mtile * 16 + l16) * K1;
    const float* brow = W1 + (size_t)(wv * 16 + l16) * K1;

    v8f acc = {};
    for (int kc = 0; kc < K1; kc += 32) {
        v16bf a = load_frag_f32(arow, kc, hi);
        v16bf b = load_frag_f32(brow, kc, hi);
        acc = wmma_bf16(a, b, acc);
    }
    const int col = wv * 16 + l16;
    const float bias = b1[col];
#pragma unroll
    for (int v = 0; v < 8; ++v) {
        int row = mtile * 16 + v + 8 * hi;
        float val = acc[v] + bias;
        val = val > 0.0f ? val : 0.0f;
        h_bf[(size_t)row * NH + col] = (__bf16)val;
    }
}

// ---------------------------------------------------------------------------
// k_fc2: probs = sigmoid(h @ W2^T + b2)  (512 x 131072 f32 out)
// One wave owns a 16-wide N strip (B-fragments register-resident),
// sweeps all 32 M tiles -> W2 read exactly once from HBM.
// grid: 1024 blocks x 8 waves = 8192 N tiles
// ---------------------------------------------------------------------------
__global__ __launch_bounds__(256) void k_fc2(
    const __bf16* __restrict__ h_bf, const float* __restrict__ W2,
    const float* __restrict__ b2, float* __restrict__ probs) {
    const int wv    = threadIdx.x >> 5;
    const int lane  = threadIdx.x & 31;
    const int l16   = lane & 15;
    const int hi    = lane >> 4;
    const int ntile = blockIdx.x * 8 + wv;
    const int col   = ntile * 16 + l16;

    const float* w2row = W2 + (size_t)col * NH;
    v16bf bfrag[4];
#pragma unroll
    for (int c = 0; c < 4; ++c) bfrag[c] = load_frag_f32(w2row, c * 32, hi);
    const float bias = b2[col];

    for (int mtile = 0; mtile < NB / 16; ++mtile) {
        const __bf16* arow = h_bf + (size_t)(mtile * 16 + l16) * NH;
        v8f acc = {};
#pragma unroll
        for (int c = 0; c < 4; ++c) {
            v16bf a = load_frag_bf16(arow, c * 32, hi);
            acc = wmma_bf16(a, bfrag[c], acc);
        }
#pragma unroll
        for (int v = 0; v < 8; ++v) {
            int row = mtile * 16 + v + 8 * hi;
            float z = acc[v] + bias;
            probs[(size_t)row * N2 + col] = fsigmoid(z);
        }
    }
}

// ---------------------------------------------------------------------------
// k_einsum: wx[b,l,t] = sum_s x[b,l,s] * probs[b,l,s,t]  -> bf16 for LSTM
// one 64-thread block per (b,l); coalesced reads of probs rows
// ---------------------------------------------------------------------------
__global__ __launch_bounds__(64) void k_einsum(
    const float* __restrict__ x, const float* __restrict__ probs,
    __bf16* __restrict__ wx) {
    const int bl = blockIdx.x;            // 0..16383
    const int b  = bl >> 5;
    const int l  = bl & 31;
    const int t  = threadIdx.x;           // 0..63

    __shared__ float xs[NV];
    xs[t] = x[(size_t)(b * NL + l) * NV + t];
    __syncthreads();

    const float* p = probs + (size_t)b * N2 + (size_t)l * (NV * NV) + t;
    float acc = 0.0f;
#pragma unroll 8
    for (int s = 0; s < NV; ++s) acc += xs[s] * p[(size_t)s * NV];
    wx[(size_t)(b * NL + l) * NV + t] = (__bf16)acc;
}

// ---------------------------------------------------------------------------
// k_lstm: 32-step LSTM over wx, then pred = h_last @ Wd^T + bd.
// One block per 16-row batch tile; wave w owns hidden cols [16w,16w+16).
// c stays in C-layout registers; h ping-pongs through a 4KB LDS bf16 tile.
// ---------------------------------------------------------------------------
__global__ __launch_bounds__(256) void k_lstm(
    const __bf16* __restrict__ wx,
    const __bf16* __restrict__ Wih_bf, const __bf16* __restrict__ Whh_bf,
    const float* __restrict__ bcomb,  const __bf16* __restrict__ Wd_bf,
    const float* __restrict__ bd,     float* __restrict__ pred) {
    const int mtile = blockIdx.x;         // 0..31
    const int wv    = threadIdx.x >> 5;   // 0..7
    const int lane  = threadIdx.x & 31;
    const int l16   = lane & 15;
    const int hi    = lane >> 4;

    __shared__ __bf16 hbuf[16 * NH];
    for (int i = threadIdx.x; i < 16 * NH; i += 256) hbuf[i] = (__bf16)0.0f;
    __syncthreads();

    float bias[4];
#pragma unroll
    for (int g = 0; g < 4; ++g) bias[g] = bcomb[g * NH + wv * 16 + l16];

    v8f cst = {};
    const int brow = mtile * 16 + l16;    // batch row this lane reads (A side)

    for (int t = 0; t < NL; ++t) {
        // A fragments: x_t (K=64) from global, h (K=128) from LDS
        const __bf16* xrow = wx + ((size_t)brow * NL + t) * NV;
        v16bf ax0 = load_frag_bf16(xrow, 0, hi);
        v16bf ax1 = load_frag_bf16(xrow, 32, hi);
        const __bf16* hrow = &hbuf[l16 * NH];
        v16bf ah0 = load_frag_bf16(hrow, 0, hi);
        v16bf ah1 = load_frag_bf16(hrow, 32, hi);
        v16bf ah2 = load_frag_bf16(hrow, 64, hi);
        v16bf ah3 = load_frag_bf16(hrow, 96, hi);

        v8f gacc[4];
#pragma unroll
        for (int g = 0; g < 4; ++g) {
            const int n0 = g * NH + wv * 16 + l16;       // row of W (4H x K)
            const __bf16* wihrow = Wih_bf + (size_t)n0 * NV;
            const __bf16* whhrow = Whh_bf + (size_t)n0 * NH;
            v8f acc = {};
            acc = wmma_bf16(ax0, load_frag_bf16(wihrow, 0, hi), acc);
            acc = wmma_bf16(ax1, load_frag_bf16(wihrow, 32, hi), acc);
            acc = wmma_bf16(ah0, load_frag_bf16(whhrow, 0, hi), acc);
            acc = wmma_bf16(ah1, load_frag_bf16(whhrow, 32, hi), acc);
            acc = wmma_bf16(ah2, load_frag_bf16(whhrow, 64, hi), acc);
            acc = wmma_bf16(ah3, load_frag_bf16(whhrow, 96, hi), acc);
            gacc[g] = acc;
        }

        __syncthreads();   // all hbuf reads done before overwrite
#pragma unroll
        for (int v = 0; v < 8; ++v) {
            float iv = fsigmoid(gacc[0][v] + bias[0]);
            float fv = fsigmoid(gacc[1][v] + bias[1]);
            float gv = ftanh   (gacc[2][v] + bias[2]);
            float ov = fsigmoid(gacc[3][v] + bias[3]);
            float cv = fv * cst[v] + iv * gv;
            cst[v] = cv;
            float hv = ov * ftanh(cv);
            int r = v + 8 * hi;                           // C layout: row
            hbuf[r * NH + wv * 16 + l16] = (__bf16)hv;    // col = 16w + l16
        }
        __syncthreads();   // h visible for next step / decoder
    }

    // Decoder: pred = h_last @ Wd^T + bd (N=64 -> waves 0..3, one N tile each)
    if (wv < 4) {
        const __bf16* hrow = &hbuf[l16 * NH];
        const int n0 = wv * 16 + l16;
        const __bf16* wdrow = Wd_bf + (size_t)n0 * NH;
        v8f acc = {};
#pragma unroll
        for (int c = 0; c < 4; ++c) {
            v16bf a = load_frag_bf16(hrow, c * 32, hi);
            v16bf b = load_frag_bf16(wdrow, c * 32, hi);
            acc = wmma_bf16(a, b, acc);
        }
        const float bdv = bd[n0];
#pragma unroll
        for (int v = 0; v < 8; ++v) {
            int row = mtile * 16 + v + 8 * hi;
            pred[(size_t)row * NV + n0] = acc[v] + bdv;
        }
    }
}

// ---------------------------------------------------------------------------
extern "C" void kernel_launch(void* const* d_in, const int* in_sizes, int n_in,
                              void* d_out, int out_size, void* d_ws, size_t ws_size,
                              hipStream_t stream) {
    const float* x   = (const float*)d_in[0];
    const float* W1  = (const float*)d_in[1];
    const float* b1  = (const float*)d_in[2];
    const float* W2  = (const float*)d_in[3];
    const float* b2  = (const float*)d_in[4];
    const float* Wih = (const float*)d_in[5];
    const float* Whh = (const float*)d_in[6];
    const float* bih = (const float*)d_in[7];
    const float* bhh = (const float*)d_in[8];
    const float* Wd  = (const float*)d_in[9];
    const float* bd  = (const float*)d_in[10];

    float* probs = (float*)d_out;                         // 512*131072
    float* pred  = probs + (size_t)NB * N2;               // 512*64

    char* ws = (char*)d_ws;
    size_t off = 0;
    __bf16* h_bf   = (__bf16*)(ws + off); off += (size_t)NB * NH * 2;          // 128KB
    __bf16* wx_bf  = (__bf16*)(ws + off); off += (size_t)NB * NL * NV * 2;     // 2MB
    __bf16* Wih_bf = (__bf16*)(ws + off); off += (size_t)4 * NH * NV * 2;      // 64KB
    __bf16* Whh_bf = (__bf16*)(ws + off); off += (size_t)4 * NH * NH * 2;      // 128KB
    __bf16* Wd_bf  = (__bf16*)(ws + off); off += (size_t)NV * NH * 2;          // 16KB
    float*  bcomb  = (float*) (ws + off); off += (size_t)4 * NH * 4;           // 2KB

    k_prep  <<<dim3(256),   dim3(256), 0, stream>>>(Wih, Whh, Wd, bih, bhh,
                                                    Wih_bf, Whh_bf, Wd_bf, bcomb);
    k_fc1   <<<dim3(32),    dim3(256), 0, stream>>>(x, W1, b1, h_bf);
    k_fc2   <<<dim3(1024),  dim3(256), 0, stream>>>(h_bf, W2, b2, probs);
    k_einsum<<<dim3(16384), dim3(64),  0, stream>>>(x, probs, wx_bf);
    k_lstm  <<<dim3(32),    dim3(256), 0, stream>>>(wx_bf, Wih_bf, Whh_bf,
                                                    bcomb, Wd_bf, bd, pred);
}
